// QRNNLayer_12300786335949
// MI455X (gfx1250) — matrix operand order repaired and checked by reading, not compiled, not run on hardware
//
#include <hip/hip_runtime.h>
#include <hip/hip_bf16.h>
#include <math.h>

// Problem constants (match reference)
#define QB   16
#define QS   2048
#define QD   512
#define QH   512
#define Q3H  1536
#define Q2D  1024          // K-dim of the conv-as-GEMM (2 taps * D)
#define QSP  (QS + 1)      // padded time rows in xb (row 0 = zero guard)
#define QNC  32            // scan chunks
#define QL   64            // scan chunk length (QNC*QL == QS)
#define QBSH ((size_t)QB * QS * QH)   // 16777216

typedef __attribute__((ext_vector_type(16))) __bf16 bf16x16;
typedef __attribute__((ext_vector_type(8)))  __bf16 bf16x8;
typedef __attribute__((ext_vector_type(8)))  float  f32x8;

union ABFrag { bf16x16 v; bf16x8 h[2]; };

__device__ __forceinline__ unsigned short f2bf(float f) {
    unsigned int u = __float_as_uint(f);
    u += 0x7FFFu + ((u >> 16) & 1u);    // round to nearest even
    return (unsigned short)(u >> 16);
}

// ---------------- Pass 0a: x fp32 -> bf16 with zero guard row per batch ----------------
// xb layout: [B][S+1][D]; row 0 is zeros (the causal left-pad), row 1+t = x[b,t]
__global__ __launch_bounds__(256) void qrnn_cvt_x(const float* __restrict__ x,
                                                  unsigned short* __restrict__ xb) {
    int i = blockIdx.x * blockDim.x + threadIdx.x;   // over B*(S+1)*D/4
    int d4 = i % (QD / 4);
    int r  = (i / (QD / 4)) % QSP;
    int b  = i / ((QD / 4) * QSP);
    ushort4 o = {0, 0, 0, 0};
    if (r > 0) {
        const float4 v = ((const float4*)x)[((size_t)b * QS + (r - 1)) * (QD / 4) + d4];
        o.x = f2bf(v.x); o.y = f2bf(v.y); o.z = f2bf(v.z); o.w = f2bf(v.w);
    }
    ((ushort4*)xb)[i] = o;
}

// ---------------- Pass 0b: conv_w [3H,D,K] fp32 -> wt [3H, 2D] bf16 ----------------
// wt[o][kdim], kdim = k*D + d  (so A row = [x_{t-1}, x_t] matches)
__global__ __launch_bounds__(256) void qrnn_cvt_w(const float* __restrict__ w,
                                                  unsigned short* __restrict__ wt) {
    int j = blockIdx.x * blockDim.x + threadIdx.x;   // over 3H*2D
    if (j >= Q3H * Q2D) return;
    int o  = j / Q2D;
    int kd = j % Q2D;
    int k  = kd / QD;
    int d  = kd % QD;
    wt[j] = f2bf(w[(o * QD + d) * 2 + k]);
}

// ---------------- Pass 1: conv-as-GEMM + activations (WMMA bf16) ----------------
// grid = (S/128, 3H/128, B), block = 256 (8 waves). Block tile: 128 (time) x 128 (chan).
// Wave (wm 0..3, ng 0..1) computes rows [wm*32,+32) x cols [ng*64,+64) = 8 WMMAs/k-step.
// Double-buffered LDS, one barrier per k-step.
__global__ __launch_bounds__(256) void qrnn_gemm(const unsigned short* __restrict__ xb,
                                                 const unsigned short* __restrict__ wt,
                                                 const float* __restrict__ bias,
                                                 float* __restrict__ gates) {
    __shared__ unsigned short As[2][128][40];   // 80B row pitch: bank-spread, 16B-aligned cols
    __shared__ unsigned short Bs[2][128][40];

    const int m0 = blockIdx.x * 128;
    const int n0 = blockIdx.y * 128;
    const int b  = blockIdx.z;

    const int lane = threadIdx.x & 31;
    const int wave = threadIdx.x >> 5;
    const int wm = wave & 3;
    const int ng = wave >> 2;
    const int ln = lane & 15;
    const int lh = lane >> 4;

    f32x8 acc[2][4] = {};

    // cooperative-load mapping: each thread moves 16 A elems + 16 B elems per tile
    const int lr = threadIdx.x >> 1;          // 0..127 (tile row)
    const int lc = (threadIdx.x & 1) * 16;    // 0 or 16 (tile col base)
    const size_t xbase = (size_t)b * QSP * QD;             // padded batch base
    const size_t wbase = (size_t)(n0 + lr) * Q2D + lc;     // weight row base

    bf16x8 a0, a1, b0, b1;   // staged global data for the *next* tile

    auto load_tile = [&](int kk) {
        // A: tap0 (kk<D) reads padded row m0+lr (== x_{t-1}); tap1 reads m0+lr+1 (== x_t)
        const int prow = m0 + lr + (kk >= QD);
        const int col  = (kk & (QD - 1)) + lc;
        const unsigned short* asrc = &xb[xbase + (size_t)prow * QD + col];
        a0 = *(const bf16x8*)&asrc[0];
        a1 = *(const bf16x8*)&asrc[8];
        const unsigned short* wsrc = &wt[wbase + kk];
        b0 = *(const bf16x8*)&wsrc[0];
        b1 = *(const bf16x8*)&wsrc[8];
        if (kk + 32 < Q2D) {
            __builtin_prefetch(&wsrc[Q2D >= kk + 32 ? 32 : 0], 0, 3);
            __builtin_prefetch(&asrc[32], 0, 3);
        }
    };
    auto store_tile = [&](int buf) {
        *(bf16x8*)&As[buf][lr][lc]     = a0;
        *(bf16x8*)&As[buf][lr][lc + 8] = a1;
        *(bf16x8*)&Bs[buf][lr][lc]     = b0;
        *(bf16x8*)&Bs[buf][lr][lc + 8] = b1;
    };
    auto compute = [&](int buf) {
        // A fragments: lane = M row, elems 0..7 = K[kb..kb+7], elems 8..15 = K[16+kb..]
        const int kb = lh * 8;
        ABFrag a[2];
        #pragma unroll
        for (int mi = 0; mi < 2; ++mi) {
            const int mrow = wm * 32 + mi * 16 + ln;
            a[mi].h[0] = *(const bf16x8*)&As[buf][mrow][kb];
            a[mi].h[1] = *(const bf16x8*)&As[buf][mrow][16 + kb];
        }
        #pragma unroll
        for (int j = 0; j < 4; ++j) {
            // B fragment: N = lane&15, elems 0..7 = K[kb..kb+7], elems 8..15 = K[16+kb..] (matches A)
            const int nrow = ng * 64 + j * 16 + ln;
            ABFrag bb;
            bb.h[0] = *(const bf16x8*)&Bs[buf][nrow][kb];
            bb.h[1] = *(const bf16x8*)&Bs[buf][nrow][16 + kb];
            acc[0][j] = __builtin_amdgcn_wmma_f32_16x16x32_bf16(
                false, a[0].v, false, bb.v, (short)0, acc[0][j], false, false);
            acc[1][j] = __builtin_amdgcn_wmma_f32_16x16x32_bf16(
                false, a[1].v, false, bb.v, (short)0, acc[1][j], false, false);
        }
    };

    // software pipeline: tile i lives in buf[i&1]; one barrier per k-step
    load_tile(0);
    store_tile(0);
    __syncthreads();
    #pragma unroll 2
    for (int i = 0; i < (Q2D / 32) - 1; ++i) {
        const int cur = i & 1;
        load_tile((i + 1) * 32);   // global loads in flight during compute
        compute(cur);
        store_tile(1 - cur);       // target buffer last read one barrier ago: safe
        __syncthreads();
    }
    compute(((Q2D / 32) - 1) & 1);

    // --- epilogue: bias + activation, write gates[B,S,3H] f32 ---
    #pragma unroll
    for (int mi = 0; mi < 2; ++mi) {
        #pragma unroll
        for (int j = 0; j < 4; ++j) {
            const int o = n0 + ng * 64 + j * 16 + ln;     // channel
            const float bia = bias[o];
            size_t gbase = ((size_t)b * QS + (m0 + wm * 32 + mi * 16 + lh * 8)) * Q3H + o;
            #pragma unroll
            for (int r = 0; r < 8; ++r) {
                float v = acc[mi][j][r] + bia;
                float act = (o < QH) ? tanhf(v)                       // Z
                                     : 1.0f / (1.0f + __expf(-v));    // F, O
                gates[gbase + (size_t)r * Q3H] = act;
            }
        }
    }
}

// ---------------- Pass 2a: per-chunk affine aggregates (Fprod, G) ----------------
__global__ __launch_bounds__(256) void qrnn_scanA(const float* __restrict__ gates,
                                                  float* __restrict__ fp,
                                                  float* __restrict__ gg) {
    int idx = blockIdx.x * blockDim.x + threadIdx.x;   // B*NC*H
    int h = idx % QH;
    int chunk = (idx / QH) % QNC;
    int b = idx / (QH * QNC);
    size_t base = ((size_t)b * QS + chunk * QL) * Q3H + h;
    float F = 1.0f, G = 0.0f;
    #pragma unroll 4
    for (int t = 0; t < QL; ++t) {
        float z = gates[base];
        float f = gates[base + QH];
        G = f * G + (1.0f - f) * z;
        F *= f;
        base += Q3H;
    }
    fp[idx] = F;
    gg[idx] = G;
}

// ---------------- Pass 2b: tiny sequential scan over chunk aggregates ----------------
__global__ __launch_bounds__(256) void qrnn_scanB(const float* __restrict__ fp,
                                                  const float* __restrict__ gg,
                                                  float* __restrict__ cstart) {
    int idx = blockIdx.x * blockDim.x + threadIdx.x;   // B*H
    int h = idx % QH;
    int b = idx / QH;
    float c = 0.0f;
    for (int chunk = 0; chunk < QNC; ++chunk) {
        int a = (b * QNC + chunk) * QH + h;
        cstart[a] = c;
        c = fp[a] * c + gg[a];
    }
}

// ---------------- Pass 2c: rescan chunks with correct carry, write C and Hout ----------------
__global__ __launch_bounds__(256) void qrnn_scanC(const float* __restrict__ gates,
                                                  const float* __restrict__ cstart,
                                                  float* __restrict__ out) {
    int idx = blockIdx.x * blockDim.x + threadIdx.x;   // B*NC*H
    int h = idx % QH;
    int chunk = (idx / QH) % QNC;
    int b = idx / (QH * QNC);
    float c = cstart[idx];
    size_t gbase = ((size_t)b * QS + chunk * QL) * Q3H + h;
    size_t obase = ((size_t)b * QS + chunk * QL) * QH + h;
    #pragma unroll 4
    for (int t = 0; t < QL; ++t) {
        float z = gates[gbase];
        float f = gates[gbase + QH];
        float o = gates[gbase + 2 * QH];
        c = f * c + (1.0f - f) * z;
        out[obase]        = c;         // C
        out[QBSH + obase] = o * c;     // Hout
        gbase += Q3H;
        obase += QH;
    }
}

extern "C" void kernel_launch(void* const* d_in, const int* in_sizes, int n_in,
                              void* d_out, int out_size, void* d_ws, size_t ws_size,
                              hipStream_t stream) {
    (void)in_sizes; (void)n_in; (void)out_size; (void)ws_size;
    const float* x      = (const float*)d_in[0];
    const float* conv_w = (const float*)d_in[1];
    const float* conv_b = (const float*)d_in[2];
    float* out = (float*)d_out;

    // workspace partition (all offsets 256B-aligned by construction)
    char* ws = (char*)d_ws;
    unsigned short* xb = (unsigned short*)ws;                    // B*(S+1)*D bf16  = 32 MiB
    ws += (size_t)QB * QSP * QD * 2;
    unsigned short* wt = (unsigned short*)ws;                    // 3H*2D bf16      = 3 MiB
    ws += (size_t)Q3H * Q2D * 2;
    float* gates = (float*)ws;                                   // B*S*3H f32      = 192 MiB
    ws += (size_t)QB * QS * Q3H * 4;
    float* fp = (float*)ws;     ws += (size_t)QB * QNC * QH * 4; // 1 MiB
    float* gg = (float*)ws;     ws += (size_t)QB * QNC * QH * 4; // 1 MiB
    float* cs = (float*)ws;                                      // 1 MiB

    // Pass 0: precision conversion (+ causal zero guard row)
    qrnn_cvt_x<<<(QB * QSP * QD / 4) / 256, 256, 0, stream>>>(x, xb);
    qrnn_cvt_w<<<(Q3H * Q2D + 255) / 256, 256, 0, stream>>>(conv_w, wt);

    // Pass 1: conv-as-GEMM via v_wmma_f32_16x16x32_bf16 + activations
    dim3 grid(QS / 128, Q3H / 128, QB);
    qrnn_gemm<<<grid, 256, 0, stream>>>(xb, wt, conv_b, gates);

    // Pass 2: chunked linear-recurrence scan
    qrnn_scanA<<<(QB * QNC * QH) / 256, 256, 0, stream>>>(gates, fp, gg);
    qrnn_scanB<<<(QB * QH) / 256, 256, 0, stream>>>(fp, gg, cs);
    qrnn_scanC<<<(QB * QNC * QH) / 256, 256, 0, stream>>>(gates, cs, out);
}
